// VectorQuantizer_56590489092795
// MI455X (gfx1250) — matrix operand order repaired and checked by reading, not compiled, not run on hardware
//
#include <hip/hip_runtime.h>
#include <hip/hip_bf16.h>

// VQ-VAE vector quantizer for MI455X (gfx1250, wave32, WMMA).
// bf16 WMMA GEMM (z @ emb^T) with fused argmin epilogue; the N x K distance
// matrix is never materialized. v2: 2 k-tiles (32 codes) per iteration with
// two independent WMMA accumulation chains + ping-pong LDS double buffering
// (one barrier per iteration, staging overlapped with compute).

#define VQ_N 32768
#define VQ_K 8192
#define VQ_D 512
#define DSTEPS 16 // VQ_D / 32 (K-dim steps of v_wmma_f32_16x16x32_bf16)

typedef __attribute__((ext_vector_type(16))) __bf16 bf16x16;
typedef __attribute__((ext_vector_type(8)))  __bf16 bf16x8;
typedef __attribute__((ext_vector_type(4)))  __bf16 bf16x4;
typedef __attribute__((ext_vector_type(8)))  float  f32x8;

// ---------------------------------------------------------------- cvt z -> bf16
__global__ __launch_bounds__(256) void vq_cvt_z(const float* __restrict__ z,
                                                __bf16* __restrict__ zb) {
    const size_t i = (size_t)blockIdx.x * 256 + threadIdx.x; // 8 elems / thread
    const float4 f0 = ((const float4*)z)[2 * i];
    const float4 f1 = ((const float4*)z)[2 * i + 1];
    bf16x8 o;
    o[0] = (__bf16)f0.x; o[1] = (__bf16)f0.y; o[2] = (__bf16)f0.z; o[3] = (__bf16)f0.w;
    o[4] = (__bf16)f1.x; o[5] = (__bf16)f1.y; o[6] = (__bf16)f1.z; o[7] = (__bf16)f1.w;
    ((bf16x8*)zb)[i] = o;
}

// ------------------------------------------- cvt emb -> bf16, fused ||e_k||^2
__global__ __launch_bounds__(256) void vq_cvt_e(const float* __restrict__ e,
                                                __bf16* __restrict__ eb,
                                                float* __restrict__ enorm) {
    const int lane = threadIdx.x & 31;
    const int row  = blockIdx.x * 8 + (threadIdx.x >> 5); // one wave per row
    const float4* src = (const float4*)(e + (size_t)row * VQ_D);
    bf16x4*       dst = (bf16x4*)(eb + (size_t)row * VQ_D);
    float sum = 0.0f;
#pragma unroll
    for (int t = 0; t < 4; ++t) {
        const float4 f = src[lane + 32 * t];
        sum = __builtin_fmaf(f.x, f.x, sum);
        sum = __builtin_fmaf(f.y, f.y, sum);
        sum = __builtin_fmaf(f.z, f.z, sum);
        sum = __builtin_fmaf(f.w, f.w, sum);
        bf16x4 o;
        o[0] = (__bf16)f.x; o[1] = (__bf16)f.y; o[2] = (__bf16)f.z; o[3] = (__bf16)f.w;
        dst[lane + 32 * t] = o;
    }
#pragma unroll
    for (int off = 16; off >= 1; off >>= 1) sum += __shfl_xor(sum, off, 32);
    if (lane == 0) enorm[row] = sum;
}

// -------------------------------------- WMMA GEMM with fused argmin epilogue
// Block = 8 waves; wave w owns z rows [blk*128 + w*16, +16). Per iteration a
// block processes 32 codes: 32 KB staged into the *other* ping-pong buffer
// while the WMMAs consume the current one; two independent accumulators.
__global__ __launch_bounds__(256) void vq_argmin(const __bf16* __restrict__ zb,
                                                 const __bf16* __restrict__ eb,
                                                 const float* __restrict__ enorm,
                                                 int* __restrict__ idx_out,
                                                 float* __restrict__ idxf_out) {
    __shared__ alignas(16) __bf16 ldsb[2][32 * VQ_D]; // 2 x 32 KB ping-pong

    const int lane   = threadIdx.x & 31;
    const int wave   = threadIdx.x >> 5;
    const int half   = lane >> 4;  // half-wave selects K-chunk group
    const int l16    = lane & 15;
    const int m_base = (blockIdx.x * 8 + wave) * 16;
    const int NT     = VQ_K / 32;  // 256 double-tiles

    // Preload A fragments (16 z-rows x 512) once, in 16-bit A-matrix layout:
    // lane%16 = row M; half 0 holds K-chunks [0..7],[16..23]; half 1 [8..15],[24..31].
    const __bf16* zrow = zb + (size_t)(m_base + l16) * VQ_D;
    bf16x16 a[DSTEPS];
#pragma unroll
    for (int s = 0; s < DSTEPS; ++s) {
        const __bf16* p  = zrow + s * 32 + half * 8;
        const bf16x8  lo = *(const bf16x8*)(p);
        const bf16x8  hi = *(const bf16x8*)(p + 16);
        a[s] = __builtin_shufflevector(lo, hi, 0,1,2,3,4,5,6,7,8,9,10,11,12,13,14,15);
    }

    float bestv[8];
    int   besti[8];
#pragma unroll
    for (int j = 0; j < 8; ++j) { bestv[j] = __builtin_inff(); besti[j] = 0; }

    // Prologue: stage double-tile 0 into buffer 0 (32 KB = 2048 uint4).
    {
        const uint4* src = (const uint4*)eb;
        uint4*       dst = (uint4*)ldsb[0];
#pragma unroll
        for (int i = 0; i < 8; ++i)
            dst[threadIdx.x + 256 * i] = src[threadIdx.x + 256 * i];
    }
    __syncthreads();

    for (int t = 0; t < NT; ++t) {
        const __bf16* cur = ldsb[t & 1];

        // Stage next double-tile into the other buffer; overlaps with WMMAs
        // below (independent: scheduler hoists loads, ds_stores complete
        // before the end-of-iteration barrier).
        if (t + 1 < NT) {
            const uint4* src = (const uint4*)(eb + (size_t)(t + 1) * 32 * VQ_D);
            uint4*       dst = (uint4*)ldsb[(t + 1) & 1];
#pragma unroll
            for (int i = 0; i < 8; ++i)
                dst[threadIdx.x + 256 * i] = src[threadIdx.x + 256 * i];
            if (t + 2 < NT) // prefetch tile t+2 (global_prefetch_b8)
                __builtin_prefetch((const char*)(src + 2048) + (size_t)threadIdx.x * 128, 0, 1);
        }

        // Two 16x16 dot tiles: codes [t*32, t*32+16) and [t*32+16, t*32+32).
        // B-matrix (32x16) layout: lane%16 = column N (code); half-wave 0
        // holds K=[0..7],[16..23], half-wave 1 holds K=[8..15],[24..31].
        f32x8 c0 = {}, c1 = {};
        const __bf16* brow0 = cur + (size_t)l16 * VQ_D + half * 8;
        const __bf16* brow1 = brow0 + (size_t)16 * VQ_D;
#pragma unroll
        for (int s = 0; s < DSTEPS; ++s) {
            const bf16x8 lo0 = *(const bf16x8*)(brow0 + s * 32);
            const bf16x8 hi0 = *(const bf16x8*)(brow0 + s * 32 + 16);
            const bf16x8 lo1 = *(const bf16x8*)(brow1 + s * 32);
            const bf16x8 hi1 = *(const bf16x8*)(brow1 + s * 32 + 16);
            const bf16x16 b0 =
                __builtin_shufflevector(lo0, hi0, 0,1,2,3,4,5,6,7,8,9,10,11,12,13,14,15);
            const bf16x16 b1 =
                __builtin_shufflevector(lo1, hi1, 0,1,2,3,4,5,6,7,8,9,10,11,12,13,14,15);
            // Two independent accumulation chains sharing the A fragment.
            c0 = __builtin_amdgcn_wmma_f32_16x16x32_bf16(
                     false, a[s], false, b0, (short)0, c0, false, false);
            c1 = __builtin_amdgcn_wmma_f32_16x16x32_bf16(
                     false, a[s], false, b1, (short)0, c1, false, false);
        }

        // Epilogue: score = ||e_k||^2 - 2*dot (||z||^2 constant per row).
        const int   k0  = t * 32 + l16;
        const int   k1  = k0 + 16;
        const float en0 = enorm[k0];
        const float en1 = enorm[k1];
#pragma unroll
        for (int j = 0; j < 8; ++j) {
            const float s0 = __builtin_fmaf(-2.0f, c0[j], en0);
            const float s1 = __builtin_fmaf(-2.0f, c1[j], en1);
            if (s0 < bestv[j]) { bestv[j] = s0; besti[j] = k0; } // k ascending ->
            if (s1 < bestv[j]) { bestv[j] = s1; besti[j] = k1; } // first-min wins
        }

        __syncthreads(); // next buffer fully staged; cur free for overwrite
    }

    // Cross-lane argmin within each 16-lane half. C layout: lanes 0-15 hold
    // rows m_base+0..7 (vgpr j -> row j), lanes 16-31 hold rows m_base+8..15.
#pragma unroll
    for (int off = 8; off >= 1; off >>= 1) {
#pragma unroll
        for (int j = 0; j < 8; ++j) {
            const float ov = __shfl_xor(bestv[j], off, 16);
            const int   oi = __shfl_xor(besti[j], off, 16);
            if (ov < bestv[j] || (ov == bestv[j] && oi < besti[j])) {
                bestv[j] = ov; besti[j] = oi;
            }
        }
    }
    if (l16 == 0) {
#pragma unroll
        for (int j = 0; j < 8; ++j) {
            const int row = m_base + half * 8 + j;
            idx_out[row]  = besti[j];
            idxf_out[row] = (float)besti[j];
        }
    }
}

// -------------------- gather fp32 codebook rows + deterministic partial loss
__global__ __launch_bounds__(128) void vq_gather(const float* __restrict__ emb,
                                                 const float* __restrict__ z,
                                                 const int* __restrict__ idx,
                                                 float* __restrict__ out_q,
                                                 float* __restrict__ partial) {
    __shared__ float red[4];
    const int row = blockIdx.x;
    const int k   = idx[row];
    const float4* e4 = (const float4*)(emb + (size_t)k * VQ_D);
    const float4* z4 = (const float4*)(z + (size_t)row * VQ_D);
    float4*       q4 = (float4*)(out_q + (size_t)row * VQ_D);
    const int t = threadIdx.x; // 128 threads x float4 = 512 elems
    const float4 ev = e4[t];
    const float4 zv = z4[t];
    q4[t] = ev; // quantized_st forward value == quantized
    const float dx = ev.x - zv.x, dy = ev.y - zv.y;
    const float dz = ev.z - zv.z, dw = ev.w - zv.w;
    float s = dx * dx + dy * dy + dz * dz + dw * dw;
#pragma unroll
    for (int off = 16; off >= 1; off >>= 1) s += __shfl_xor(s, off, 32);
    if ((t & 31) == 0) red[t >> 5] = s;
    __syncthreads();
    if (t == 0) partial[row] = red[0] + red[1] + red[2] + red[3];
}

// ------------------------------- deterministic final reduction: 1.25 * mean
__global__ __launch_bounds__(256) void vq_finalize(const float* __restrict__ partial,
                                                   float* __restrict__ loss_out) {
    __shared__ float sm[256];
    float s = 0.0f;
    for (int i = threadIdx.x; i < VQ_N; i += 256) s += partial[i]; // fixed order
    sm[threadIdx.x] = s;
    __syncthreads();
    for (int off = 128; off >= 1; off >>= 1) {
        if (threadIdx.x < off) sm[threadIdx.x] += sm[threadIdx.x + off];
        __syncthreads();
    }
    if (threadIdx.x == 0)
        *loss_out = 1.25f * sm[0] / (float)((size_t)VQ_N * VQ_D);
}

extern "C" void kernel_launch(void* const* d_in, const int* in_sizes, int n_in,
                              void* d_out, int out_size, void* d_ws, size_t ws_size,
                              hipStream_t stream) {
    (void)in_sizes; (void)n_in; (void)out_size; (void)ws_size;
    const float* z = (const float*)d_in[0];
    const float* e = (const float*)d_in[1];

    // d_out layout (reference return order): quantized_st [N*D] | loss [1] | indices [N]
    float* out_q    = (float*)d_out;
    float* out_loss = out_q + (size_t)VQ_N * VQ_D;
    float* out_idx  = out_loss + 1;

    // Workspace layout (~42.3 MB)
    char*   ws      = (char*)d_ws;
    __bf16* zb      = (__bf16*)(ws);                    // 32 MB
    __bf16* eb      = (__bf16*)(ws + 33554432);         //  8 MB
    float*  enorm   = (float*) (ws + 41943040);         // 32 KB
    int*    idx     = (int*)   (ws + 41975808);         // 128 KB
    float*  partial = (float*) (ws + 42106880);         // 128 KB

    vq_cvt_z  <<<(VQ_N * VQ_D / 8) / 256, 256, 0, stream>>>(z, zb);
    vq_cvt_e  <<<VQ_K / 8,               256, 0, stream>>>(e, eb, enorm);
    vq_argmin <<<VQ_N / 128,             256, 0, stream>>>(zb, eb, enorm, idx, out_idx);
    vq_gather <<<VQ_N,                   128, 0, stream>>>(e, z, idx, out_q, partial);
    vq_finalize<<<1,                     256, 0, stream>>>(partial, out_loss);
}
